// CausalSelfAttention_23837068493144
// MI455X (gfx1250) — matrix operand order repaired and checked by reading, not compiled, not run on hardware
//
#include <hip/hip_runtime.h>

// ---------------------------------------------------------------------------
// CDNA5 (gfx1250, wave32) causal self-attention forward.
// bf16 WMMA everywhere; f32 accumulation; whole working set is L2-resident.
// GEMM tiles staged with GLOBAL_LOAD_ASYNC_TO_LDS_B128 (ASYNCcnt path),
// double-buffered in LDS.
// ---------------------------------------------------------------------------

typedef __bf16 bf16;
typedef __attribute__((ext_vector_type(8)))  bf16  v8bf;
typedef __attribute__((ext_vector_type(16))) bf16  v16bf;
typedef __attribute__((ext_vector_type(8)))  float v8f;

constexpr int kBatch = 2;
constexpr int kSeq   = 2048;
constexpr int kDim   = 2048;
constexpr int kNH    = 16;
constexpr int kNKV   = 4;
constexpr int kHD    = 128;          // kDim / kNH
constexpr int kKVD   = kNKV * kHD;   // 512
constexpr int kTok   = kBatch * kSeq;

static __device__ __forceinline__ v16bf cat16(v8bf lo, v8bf hi) {
  return __builtin_shufflevector(lo, hi, 0,1,2,3,4,5,6,7,8,9,10,11,12,13,14,15);
}

static __device__ __forceinline__ v8f wmma_bf16(v16bf a, v16bf b, v8f c) {
  // v_wmma_f32_16x16x32_bf16  (8-arg form, probe-verified)
  return __builtin_amdgcn_wmma_f32_16x16x32_bf16(
      /*neg_a=*/false, a, /*neg_b=*/false, b,
      /*c_mod=*/(short)0, c, /*reuse_a=*/false, /*reuse_b=*/false);
}

// --- CDNA5 async global->LDS copy (tracked by ASYNCcnt) --------------------
static __device__ __forceinline__ unsigned lds_addr32(const void* p) {
  // __shared__ pointers are generic in HIP; addrspacecast to AS(3) yields the
  // 32-bit LDS byte offset the VDST operand of the async op expects.
  return (unsigned)(uintptr_t)(__attribute__((address_space(3))) const char*)p;
}

static __device__ __forceinline__ void async_copy_b128(void* lds_dst,
                                                       const void* gsrc) {
  asm volatile("global_load_async_to_lds_b128 %0, %1, off"
               :
               : "v"(lds_addr32(lds_dst)),
                 "v"((unsigned long long)(uintptr_t)gsrc)
               : "memory");
}

static __device__ __forceinline__ void wait_async0() {
  asm volatile("s_wait_asynccnt 0x0" ::: "memory");
}

// ---------------------------------------------------------------------------
// f32 -> bf16 conversion (streaming; negligible vs. GEMM cost)
// ---------------------------------------------------------------------------
__global__ void f32_to_bf16(const float* __restrict__ in,
                            bf16* __restrict__ out, int n) {
  int i = blockIdx.x * blockDim.x + threadIdx.x;
  if (i < n) out[i] = (bf16)in[i];
}

// ---------------------------------------------------------------------------
// GEMM: C[M,N] = A[M,K] * W[N,K]^T  (both K-contiguous, bf16, f32 accum)
// Block 256 threads = 8 waves; block tile 128x64; BK=32; wave tile 32x32.
// LDS rows padded to 40 bf16 (80B) -> 16B-aligned ds_load_b128, low conflicts.
// Tiles double-buffered, filled by the async-to-LDS engine.
// ---------------------------------------------------------------------------
constexpr int BM = 128, BN = 64, BK = 32, LDA = 40;

template <bool OUT_F32>
__global__ __launch_bounds__(256, 2)
void gemm_xwt(const bf16* __restrict__ A, const bf16* __restrict__ W,
              void* __restrict__ Cout, int M, int N, int K)
{
  __shared__ __align__(16) bf16 As[2][BM * LDA];
  __shared__ __align__(16) bf16 Bs[2][BN * LDA];

  const int tid  = threadIdx.x;
  const int lane = tid & 31;
  const int hl   = lane >> 4;    // which 16-lane half
  const int l16  = lane & 15;
  const int wave = tid >> 5;     // 0..7
  const int wm   = wave & 3;     // 4 wave-rows of 32
  const int wn   = wave >> 2;    // 2 wave-cols of 32

  const int m0 = blockIdx.y * BM;
  const int n0 = blockIdx.x * BN;

  // Per-thread staging coordinates:
  //   A tile 128x32: thread stages 32B of row (tid>>1) at col (tid&1)*16.
  //   W tile  64x32: thread stages 16B of row (tid>>2) at col (tid&3)*8.
  const int arow = tid >> 1, acol = (tid & 1) * 16;
  const int wrow = tid >> 2, wcol = (tid & 3) * 8;

  auto stage = [&](int k0, int bi) {
    const bf16* gA = &A[(size_t)(m0 + arow) * K + k0 + acol];
    async_copy_b128(&As[bi][arow * LDA + acol],     gA);
    async_copy_b128(&As[bi][arow * LDA + acol + 8], gA + 8);
    async_copy_b128(&Bs[bi][wrow * LDA + wcol],
                    &W[(size_t)(n0 + wrow) * K + k0 + wcol]);
  };

  v8f acc[2][2];
#pragma unroll
  for (int i = 0; i < 2; ++i)
#pragma unroll
    for (int j = 0; j < 2; ++j)
      acc[i][j] = (v8f){0,0,0,0,0,0,0,0};

  const int nk = K / BK;
  stage(0, 0);

  for (int it = 0; it < nk; ++it) {
    wait_async0();        // my async stores for buffer it&1 have landed
    __syncthreads();      // ... and everyone else's too

    if (it + 1 < nk) stage((it + 1) * BK, (it + 1) & 1);
    if (it + 2 < nk)      // warm L2/WGP$ ahead of the async engine
      __builtin_prefetch(&A[(size_t)(m0 + arow) * K + (it + 2) * BK + acol], 0, 0);

    const bf16* as = As[it & 1];
    const bf16* bs = Bs[it & 1];

#pragma unroll
    for (int tm = 0; tm < 2; ++tm) {
      int ar = wm * 32 + tm * 16 + l16;
      // A-frag: K = {hl*16 .. +16} contiguous
      v16bf a = cat16(*(const v8bf*)&as[ar * LDA + hl * 16],
                      *(const v8bf*)&as[ar * LDA + hl * 16 + 8]);
#pragma unroll
      for (int tn = 0; tn < 2; ++tn) {
        int br = wn * 32 + tn * 16 + l16;
        // B-frag: lane half selects K = 0..15 / 16..31, contiguous 16 bf16
        v16bf bfr = cat16(*(const v8bf*)&bs[br * LDA + hl * 16],
                          *(const v8bf*)&bs[br * LDA + hl * 16 + 8]);
        acc[tm][tn] = wmma_bf16(a, bfr, acc[tm][tn]);
      }
    }
    __syncthreads();      // nobody overwrites buffer it&1 until all have read
  }

  // C layout: row M = r + 8*hl, col N = l16
#pragma unroll
  for (int tm = 0; tm < 2; ++tm)
#pragma unroll
    for (int tn = 0; tn < 2; ++tn)
#pragma unroll
      for (int r = 0; r < 8; ++r) {
        int grow = m0 + wm * 32 + tm * 16 + r + hl * 8;
        int gcol = n0 + wn * 32 + tn * 16 + l16;
        float v = acc[tm][tn][r];
        if constexpr (OUT_F32)
          ((float*)Cout)[(size_t)grow * N + gcol] = v;
        else
          ((bf16*)Cout)[(size_t)grow * N + gcol] = (bf16)v;
      }
}

// ---------------------------------------------------------------------------
// Fused RMS-norm (over HD=128) + RoPE + optional per-head gain, in place.
// One wave per (token, head) row; lane covers elements {j, j+32, j+64, j+96}.
// ---------------------------------------------------------------------------
__global__ __launch_bounds__(128)
void rms_rope(bf16* __restrict__ buf, const float* __restrict__ gain,
              int nheads, int rowdim, int totrows)
{
  int wid = blockIdx.x * 4 + (threadIdx.x >> 5);
  if (wid >= totrows) return;
  int lane  = threadIdx.x & 31;
  int token = wid / nheads;
  int h     = wid - token * nheads;
  int pos   = token & (kSeq - 1);
  bf16* p = buf + (size_t)token * rowdim + h * kHD;

  int j0 = lane, j1 = lane + 32;
  float x1a = (float)p[j0],  x2a = (float)p[j0 + 64];
  float x1b = (float)p[j1],  x2b = (float)p[j1 + 64];
  float ss = x1a * x1a + x2a * x2a + x1b * x1b + x2b * x2b;
#pragma unroll
  for (int m = 1; m < 32; m <<= 1) ss += __shfl_xor(ss, m, 32);
  float rn = rsqrtf(ss * (1.0f / 128.0f) + 1.1920929e-7f);
  float g  = gain ? gain[h] : 1.0f;

  const float kL2Base = 13.287712379549449f;  // log2(10000)
  float fa = exp2f(-(float)j0 * (kL2Base / 64.0f));
  float fb = exp2f(-(float)j1 * (kL2Base / 64.0f));
  float aa = (float)pos * fa, ab = (float)pos * fb;
  float ca = __cosf(aa), sa = __sinf(aa);
  float cb = __cosf(ab), sb = __sinf(ab);
  x1a *= rn; x2a *= rn; x1b *= rn; x2b *= rn;
  p[j0]      = (bf16)(g * ( x1a * ca + x2a * sa));
  p[j0 + 64] = (bf16)(g * (-x1a * sa + x2a * ca));
  p[j1]      = (bf16)(g * ( x1b * cb + x2b * sb));
  p[j1 + 64] = (bf16)(g * (-x1b * sb + x2b * cb));
}

// ---------------------------------------------------------------------------
// V transpose: vt[(b,kv,d), s] = v[(b,s), kv*HD + d]  (so PV B-frags are
// contiguous 16-bf16 loads along the key dimension).
// ---------------------------------------------------------------------------
__global__ void vtrans(const bf16* __restrict__ v, bf16* __restrict__ vt) {
  int idx = blockIdx.x * blockDim.x + threadIdx.x;
  if (idx >= kBatch * kNKV * kHD * kSeq) return;
  int s  = idx & (kSeq - 1);
  int t  = idx >> 11;            // / kSeq
  int d  = t & (kHD - 1);
  int t2 = t >> 7;               // / kHD
  int kv = t2 & (kNKV - 1);
  int b  = t2 >> 2;
  vt[idx] = v[(size_t)(b * kSeq + s) * kKVD + kv * kHD + d];
}

// ---------------------------------------------------------------------------
// Flash attention: one wave owns 16 query rows (Q frags live in registers).
// Key tiles of 32; S = Q*K^T via 8 WMMAs; online softmax in f32 with per-row
// m/l kept in the C-matrix lane layout; P restaged via LDS into A-frag layout;
// y += P*V via 8 WMMAs (vt gives contiguous key-major B-frags).
// ---------------------------------------------------------------------------
__global__ __launch_bounds__(128)
void flash_attn(const bf16* __restrict__ q, const bf16* __restrict__ kc,
                const bf16* __restrict__ vt, bf16* __restrict__ y)
{
  __shared__ __align__(16) bf16 pbuf[4][16 * LDA];

  const int wave = threadIdx.x >> 5;
  const int lane = threadIdx.x & 31;
  const int hl   = lane >> 4;
  const int l16  = lane & 15;
  const int qt = blockIdx.x;          // query tile of 64
  const int h  = blockIdx.y;
  const int b  = blockIdx.z;
  const int hk = h >> 2;              // GQA: 4 query heads per kv head
  const int q0 = qt * 64 + wave * 16; // base query position of this wave

  // Q fragments (16 rows x 128 dims = 4 k-slices of 32), register-resident.
  v16bf afr[4];
  const bf16* qrow = q + (size_t)(b * kSeq + q0 + l16) * kDim + h * kHD;
#pragma unroll
  for (int ks = 0; ks < 4; ++ks) {
    const bf16* s = qrow + ks * 32 + hl * 16;
    afr[ks] = cat16(*(const v8bf*)s, *(const v8bf*)(s + 8));
  }

  float mrow[8], lrow[8];
  v8f acc[8];
#pragma unroll
  for (int r = 0; r < 8; ++r) { mrow[r] = -1e30f; lrow[r] = 0.0f; }
#pragma unroll
  for (int t = 0; t < 8; ++t) acc[t] = (v8f){0,0,0,0,0,0,0,0};

  const float scale = 0.08838834764831845f;  // 1/sqrt(128)
  const int nkt = (q0 + 16 + 31) >> 5;       // causal tile count

  for (int kt = 0; kt < nkt; ++kt) {
    const int kb = kt * 32;
    v8f s0 = (v8f){0,0,0,0,0,0,0,0};
    v8f s1 = (v8f){0,0,0,0,0,0,0,0};
#pragma unroll
    for (int ks = 0; ks < 4; ++ks) {
      const bf16* k0p = kc + (size_t)(b * kSeq + kb + l16) * kKVD
                           + hk * kHD + ks * 32 + hl * 16;
      const bf16* k1p = kc + (size_t)(b * kSeq + kb + 16 + l16) * kKVD
                           + hk * kHD + ks * 32 + hl * 16;
      v16bf b0 = cat16(*(const v8bf*)k0p, *(const v8bf*)(k0p + 8));
      v16bf b1 = cat16(*(const v8bf*)k1p, *(const v8bf*)(k1p + 8));
      s0 = wmma_bf16(afr[ks], b0, s0);
      s1 = wmma_bf16(afr[ks], b1, s1);
    }

    float p0[8], p1[8];
#pragma unroll
    for (int r = 0; r < 8; ++r) {
      int qpos = q0 + r + hl * 8;
      float v0 = (kb + l16      <= qpos) ? s0[r] * scale : -1e30f;
      float v1 = (kb + 16 + l16 <= qpos) ? s1[r] * scale : -1e30f;
      float vm = fmaxf(v0, v1);
#pragma unroll
      for (int m = 1; m < 16; m <<= 1) vm = fmaxf(vm, __shfl_xor(vm, m, 32));
      float mnew  = fmaxf(mrow[r], vm);
      float alpha = __expf(mrow[r] - mnew);
      float e0 = __expf(v0 - mnew);
      float e1 = __expf(v1 - mnew);
      float rs = e0 + e1;
#pragma unroll
      for (int m = 1; m < 16; m <<= 1) rs += __shfl_xor(rs, m, 32);
      lrow[r] = lrow[r] * alpha + rs;
      mrow[r] = mnew;
      p0[r] = e0; p1[r] = e1;
#pragma unroll
      for (int t = 0; t < 8; ++t) acc[t][r] *= alpha;
    }

    // Restage P (16x32) from C-layout registers into A-frag layout via LDS.
#pragma unroll
    for (int r = 0; r < 8; ++r) {
      pbuf[wave][(r + hl * 8) * LDA + l16]      = (bf16)p0[r];
      pbuf[wave][(r + hl * 8) * LDA + 16 + l16] = (bf16)p1[r];
    }
    v16bf pa = cat16(*(v8bf*)&pbuf[wave][l16 * LDA + hl * 16],
                     *(v8bf*)&pbuf[wave][l16 * LDA + hl * 16 + 8]);

#pragma unroll
    for (int t = 0; t < 8; ++t) {
      const bf16* vp = vt + (size_t)((b * kNKV + hk) * kHD + t * 16 + l16) * kSeq
                          + kb + hl * 16;
      v16bf bv = cat16(*(const v8bf*)vp, *(const v8bf*)(vp + 8));
      acc[t] = wmma_bf16(pa, bv, acc[t]);
    }
  }

#pragma unroll
  for (int r = 0; r < 8; ++r) {
    float inv = 1.0f / lrow[r];
    int row = b * kSeq + q0 + r + hl * 8;
#pragma unroll
    for (int t = 0; t < 8; ++t)
      y[(size_t)row * kDim + h * kHD + t * 16 + l16] = (bf16)(acc[t][r] * inv);
  }
}

// ---------------------------------------------------------------------------
extern "C" void kernel_launch(void* const* d_in, const int* in_sizes, int n_in,
                              void* d_out, int out_size, void* d_ws, size_t ws_size,
                              hipStream_t stream)
{
  (void)in_sizes; (void)n_in; (void)out_size; (void)ws_size;

  const float* x  = (const float*)d_in[0];
  const float* wq = (const float*)d_in[1];
  const float* wk = (const float*)d_in[2];
  const float* wv = (const float*)d_in[3];
  const float* wp = (const float*)d_in[4];
  const float* qg = (const float*)d_in[5];

  bf16* ws = (bf16*)d_ws;
  size_t o = 0;
  bf16* xb  = ws + o; o += (size_t)kTok * kDim;
  bf16* wqb = ws + o; o += (size_t)kDim * kDim;
  bf16* wkb = ws + o; o += (size_t)kKVD * kDim;
  bf16* wvb = ws + o; o += (size_t)kKVD * kDim;
  bf16* wpb = ws + o; o += (size_t)kDim * kDim;
  bf16* qb  = ws + o; o += (size_t)kTok * kDim;
  bf16* kb  = ws + o; o += (size_t)kTok * kKVD;
  bf16* vb  = ws + o; o += (size_t)kTok * kKVD;
  bf16* vtb = ws + o; o += (size_t)kTok * kKVD;
  bf16* yb  = ws + o; o += (size_t)kTok * kDim;

  auto cvt = [&](const float* src, bf16* dst, int n) {
    f32_to_bf16<<<(n + 255) / 256, 256, 0, stream>>>(src, dst, n);
  };
  cvt(x,  xb,  kTok * kDim);
  cvt(wq, wqb, kDim * kDim);
  cvt(wk, wkb, kKVD * kDim);
  cvt(wv, wvb, kKVD * kDim);
  cvt(wp, wpb, kDim * kDim);

  // QKV projections (C = x * W^T), bf16 out.
  gemm_xwt<false><<<dim3(kDim / BN, kTok / BM), 256, 0, stream>>>(
      xb, wqb, qb, kTok, kDim, kDim);
  gemm_xwt<false><<<dim3(kKVD / BN, kTok / BM), 256, 0, stream>>>(
      xb, wkb, kb, kTok, kKVD, kDim);
  gemm_xwt<false><<<dim3(kKVD / BN, kTok / BM), 256, 0, stream>>>(
      xb, wvb, vb, kTok, kKVD, kDim);

  // RMS-norm + RoPE (+ gain for q).
  rms_rope<<<(kTok * kNH) / 4, 128, 0, stream>>>(qb, qg, kNH, kDim, kTok * kNH);
  rms_rope<<<(kTok * kNKV) / 4, 128, 0, stream>>>(kb, nullptr, kNKV, kKVD,
                                                  kTok * kNKV);

  vtrans<<<(kTok * kKVD + 255) / 256, 256, 0, stream>>>(vb, vtb);

  flash_attn<<<dim3(kSeq / 64, kNH, kBatch), 128, 0, stream>>>(qb, kb, vtb, yb);

  // Output projection, f32 out.
  gemm_xwt<true><<<dim3(kDim / BN, kTok / BM), 256, 0, stream>>>(
      yb, wpb, d_out, kTok, kDim, kDim);
}
